// CausalSelfAttention_78477642432843
// MI455X (gfx1250) — matrix operand order repaired and checked by reading, not compiled, not run on hardware
//
#include <hip/hip_runtime.h>
#include <hip/hip_bf16.h>
#include <math.h>

typedef __attribute__((ext_vector_type(16))) __bf16 v16bf;
typedef __attribute__((ext_vector_type(8)))  __bf16 v8bf;   // 16 bytes
typedef __attribute__((ext_vector_type(8)))  float  v8f;

#define N_EMBD   1024
#define N_HEAD   16
#define HEAD_DIM 64
#define SEQ      1024
#define BATCH    8
#define ROWS     (BATCH * SEQ)   // 8192

// ---------------------------------------------------------------- fragments

union FragU { v16bf b; v8bf h[2]; };

// A fragment (16x32 bf16, row-major source, ld in elements).
// Lane l (ISA 7.12.2): m = l&15; lanes 0-15 hold K 0-7 & 16-23, lanes 16-31 hold K 8-15 & 24-31.
__device__ __forceinline__ v16bf load_a_frag(const __bf16* base, int ld,
                                             int row0, int k0) {
  const int l  = threadIdx.x & 31;
  const int m  = l & 15;
  const int kb = (l < 16) ? 0 : 8;
  const __bf16* p = base + (size_t)(row0 + m) * ld + (k0 + kb);
  FragU f;
  f.h[0] = *reinterpret_cast<const v8bf*>(p);        // K = kb .. kb+7
  f.h[1] = *reinterpret_cast<const v8bf*>(p + 16);   // K = kb+16 .. kb+23
  return f.b;
}

// B fragment (32x16 bf16). Source is "columns-as-rows" (Wt[n][k] / K[kv][d] / Vt[d][kv]):
// lane n holds output-column n; lanes 0-15: K 0-7 & 16-23, lanes 16-31: K 8-15 & 24-31.
__device__ __forceinline__ v16bf load_b_frag(const __bf16* base, int ld,
                                             int n0, int k0) {
  const int l    = threadIdx.x & 31;
  const int n    = l & 15;
  const int koff = (l < 16) ? 0 : 8;
  const __bf16* p = base + (size_t)(n0 + n) * ld + (k0 + koff);
  FragU f;
  f.h[0] = *reinterpret_cast<const v8bf*>(p);
  f.h[1] = *reinterpret_cast<const v8bf*>(p + 16);
  return f.b;
}

__device__ __forceinline__ v8f wmma_bf16(v16bf a, v16bf b, v8f c) {
  return __builtin_amdgcn_wmma_f32_16x16x32_bf16(false, a, false, b,
                                                 (short)0, c, false, false);
}

// ---------------------------------------------------------------- async copy (CDNA5)

// Low 32 bits of a generic pointer to LDS == wave-relative LDS byte address
// (ISA 10.2: LDS aperture address is addr[31:0]).
__device__ __forceinline__ unsigned lds_addr(const void* p) {
  return (unsigned)(uintptr_t)p;
}

// 16 bytes per lane, memory -> LDS, tracked by ASYNCcnt.
__device__ __forceinline__ void async_ld16(unsigned lds_off, const __bf16* g) {
  asm volatile("global_load_async_to_lds_b128 %0, %1, off"
               :: "v"(lds_off), "v"(g) : "memory");
}

__device__ __forceinline__ void wait_async_le4() {
  asm volatile("s_wait_asynccnt 0x4" ::: "memory");
}
__device__ __forceinline__ void wait_async_0() {
  asm volatile("s_wait_asynccnt 0x0" ::: "memory");
}
// same-wave DS ops are issued & completed in order; only stop compiler motion
__device__ __forceinline__ void wave_fence() {
  asm volatile("" ::: "memory");
}

// ---------------------------------------------------------------- precision converts

// x: 8 elements per thread, 32B in / 16B out
__global__ __launch_bounds__(256) void cvt_f32_to_bf16(const float* __restrict__ in,
                                                       __bf16* __restrict__ out) {
  size_t base = ((size_t)blockIdx.x * 256 + threadIdx.x) * 8;
  float4 a = *reinterpret_cast<const float4*>(in + base);
  float4 b = *reinterpret_cast<const float4*>(in + base + 4);
  v8bf o;
  o[0] = (__bf16)a.x; o[1] = (__bf16)a.y; o[2] = (__bf16)a.z; o[3] = (__bf16)a.w;
  o[4] = (__bf16)b.x; o[5] = (__bf16)b.y; o[6] = (__bf16)b.z; o[7] = (__bf16)b.w;
  *reinterpret_cast<v8bf*>(out + base) = o;
}

// Wt[n][k] = bf16(W[k][n]), 1024x1024, LDS-tiled so both sides coalesce
__global__ __launch_bounds__(256) void transpose_cvt_w(const float* __restrict__ W,
                                                       __bf16* __restrict__ Wt) {
  __shared__ __bf16 tile[32][33];
  const int tx = threadIdx.x & 31;
  const int ty = threadIdx.x >> 5;          // 0..7
  const int n0 = blockIdx.x * 32;
  const int k0 = blockIdx.y * 32;
#pragma unroll
  for (int r = ty; r < 32; r += 8)
    tile[r][tx] = (__bf16)W[(size_t)(k0 + r) * N_EMBD + (n0 + tx)];
  __syncthreads();
#pragma unroll
  for (int r = ty; r < 32; r += 8)
    Wt[(size_t)(n0 + r) * N_EMBD + (k0 + tx)] = tile[tx][r];
}

// ---------------------------------------------------------------- 64x64 GEMM tile core

__device__ __forceinline__ void gemm_tile_64x64(const __bf16* A, int lda, int row0,
                                                const __bf16* Bt, int ldb, int col0,
                                                int K, v8f acc[4][4]) {
  for (int k0 = 0; k0 < K; k0 += 32) {
    v16bf a[4];
#pragma unroll
    for (int mi = 0; mi < 4; ++mi)
      a[mi] = load_a_frag(A, lda, row0 + 16 * mi, k0);
#pragma unroll
    for (int ni = 0; ni < 4; ++ni) {
      v16bf b = load_b_frag(Bt, ldb, col0 + 16 * ni, k0);
#pragma unroll
      for (int mi = 0; mi < 4; ++mi)
        acc[mi][ni] = wmma_bf16(a[mi], b, acc[mi][ni]);
    }
  }
}

// ---------------------------------------------------------------- fused QKV projection

__global__ __launch_bounds__(32)
void qkv_proj(const __bf16* __restrict__ xb,
              const __bf16* __restrict__ Wqt,
              const __bf16* __restrict__ Wkt,
              const __bf16* __restrict__ Wvt,
              const float* __restrict__ bq, const float* __restrict__ bk,
              const float* __restrict__ bv,
              __bf16* __restrict__ Q,   // [B,H,L,64] bf16
              __bf16* __restrict__ Kc,  // [B,H,L,64] bf16
              __bf16* __restrict__ Vt)  // [B,H,64,L] bf16
{
  const int z = blockIdx.z;
  const __bf16* Wt  = (z == 0) ? Wqt : ((z == 1) ? Wkt : Wvt);
  const float* bias = (z == 0) ? bq  : ((z == 1) ? bk  : bv);
  const int row0 = blockIdx.y * 64;
  const int col0 = blockIdx.x * 64;

  v8f acc[4][4];
#pragma unroll
  for (int mi = 0; mi < 4; ++mi)
#pragma unroll
    for (int ni = 0; ni < 4; ++ni)
#pragma unroll
      for (int j = 0; j < 8; ++j) acc[mi][ni][j] = 0.0f;

  gemm_tile_64x64(xb, N_EMBD, row0, Wt, N_EMBD, col0, N_EMBD, acc);

  const int l  = threadIdx.x & 31;
  const int n  = l & 15;
  const int hi = (l >= 16) ? 1 : 0;

  float biasv[4];
#pragma unroll
  for (int ni = 0; ni < 4; ++ni) biasv[ni] = bias[col0 + 16 * ni + n];

  if (z < 2) {
    __bf16* dst = (z == 0) ? Q : Kc;
#pragma unroll
    for (int mi = 0; mi < 4; ++mi) {
#pragma unroll
      for (int ni = 0; ni < 4; ++ni) {
#pragma unroll
        for (int j = 0; j < 8; ++j) {
          int r = row0 + 16 * mi + j + 8 * hi;   // b*1024 + l
          int c = col0 + 16 * ni + n;            // h*64 + d
          int b = r >> 10, li = r & 1023, hh = c >> 6, d = c & 63;
          size_t bh = (size_t)(b * N_HEAD + hh);
          dst[(bh * SEQ + li) * HEAD_DIM + d] = (__bf16)(acc[mi][ni][j] + biasv[ni]);
        }
      }
    }
  } else {
#pragma unroll
    for (int mi = 0; mi < 4; ++mi) {
#pragma unroll
      for (int ni = 0; ni < 4; ++ni) {
#pragma unroll
        for (int j = 0; j < 8; ++j) {
          int r = row0 + 16 * mi + j + 8 * hi;
          int c = col0 + 16 * ni + n;
          int b = r >> 10, li = r & 1023, hh = c >> 6, d = c & 63;
          size_t bh = (size_t)(b * N_HEAD + hh);
          Vt[(bh * HEAD_DIM + d) * SEQ + li] = (__bf16)(acc[mi][ni][j] + biasv[ni]);
        }
      }
    }
  }
}

// ---------------------------------------------------------------- flash attention
// 4 waves / workgroup, 128 q rows of one (b,h). Each 32-kv block of K and V is
// async-copied to LDS once per WG (GLOBAL_LOAD_ASYNC_TO_LDS_B128, ASYNCcnt
// double-buffered) and consumed by all 4 waves.

#define FA_WAVES 4

__global__ __launch_bounds__(32 * FA_WAVES)
void flash_attn(const __bf16* __restrict__ Q,
                const __bf16* __restrict__ Kc,
                const __bf16* __restrict__ Vt,
                __bf16* __restrict__ attn)  // [B,L,1024] bf16
{
  const int bh   = blockIdx.y;              // b*16 + h
  const int q0wg = blockIdx.x * (32 * FA_WAVES);
  const int tid  = threadIdx.x;
  const int w    = tid >> 5;                // wave in WG
  const int q0   = q0wg + 32 * w;           // this wave's 32 q rows (32-aligned)

  const __bf16* Qh = Q  + (size_t)bh * SEQ * HEAD_DIM;
  const __bf16* Kh = Kc + (size_t)bh * SEQ * HEAD_DIM;
  const __bf16* Vh = Vt + (size_t)bh * HEAD_DIM * SEQ;

  const int l  = tid & 31;
  const int n  = l & 15;
  const int hi = (l >= 16) ? 1 : 0;
  const int m  = l & 15;
  const int kb = hi ? 8 : 0;

  __shared__ __align__(16) __bf16 Kl[2][32 * HEAD_DIM];   // [buf][kv][d]
  __shared__ __align__(16) __bf16 Vl[2][HEAD_DIM * 32];   // [buf][d][kv]
  __shared__ __align__(16) __bf16 Pl[FA_WAVES][16 * 32];  // per-wave P bounce

  v16bf qa[2][2];                     // [qtile][dchunk]
#pragma unroll
  for (int t = 0; t < 2; ++t)
#pragma unroll
    for (int c = 0; c < 2; ++c)
      qa[t][c] = load_a_frag(Qh, HEAD_DIM, q0 + 16 * t, 32 * c);

  v8f acc[2][4];
#pragma unroll
  for (int t = 0; t < 2; ++t)
#pragma unroll
    for (int c = 0; c < 4; ++c)
#pragma unroll
      for (int j = 0; j < 8; ++j) acc[t][c][j] = 0.0f;

  float rmax[2][8], rsum[2][8];
#pragma unroll
  for (int t = 0; t < 2; ++t)
#pragma unroll
    for (int j = 0; j < 8; ++j) { rmax[t][j] = -INFINITY; rsum[t][j] = 0.0f; }

  const float scale = 0.125f;         // 1/sqrt(64)
  const int nblk = (q0wg + 32 * FA_WAVES) / 32;

  // stage kv block ib into buffer (4 x b128 async per thread: 2 K + 2 V)
  auto stage = [&](int ib) {
    const int buf = ib & 1;
    const int kv0 = ib * 32;
#pragma unroll
    for (int i = 0; i < 2; ++i) {
      int c = tid + 128 * i;                  // 0..255 : K = 32 rows x 8 chunks
      int row = c >> 3, c8 = c & 7;
      async_ld16(lds_addr(&Kl[buf][row * HEAD_DIM + c8 * 8]),
                 Kh + (size_t)(kv0 + row) * HEAD_DIM + c8 * 8);
    }
#pragma unroll
    for (int i = 0; i < 2; ++i) {
      int c = tid + 128 * i;                  // 0..255 : V = 64 rows x 4 chunks
      int d = c >> 2, c4 = c & 3;
      async_ld16(lds_addr(&Vl[buf][d * 32 + c4 * 8]),
                 Vh + (size_t)d * SEQ + kv0 + c4 * 8);
    }
  };

  stage(0);

  for (int ib = 0; ib < nblk; ++ib) {
    const int buf = ib & 1;
    const int kv0 = ib * 32;

    if (ib + 1 < nblk) { stage(ib + 1); wait_async_le4(); }  // in-order: first 4 done
    else               { wait_async_0(); }
    __syncthreads();                        // block ib visible to all waves

    if (kv0 < q0 + 32) {                    // wave-uniform causal skip
      const __bf16* Kb = &Kl[buf][0];
      const __bf16* Vb = &Vl[buf][0];
      v16bf kb0 = load_b_frag(Kb, HEAD_DIM, 0,  0);
      v16bf kb1 = load_b_frag(Kb, HEAD_DIM, 0,  32);
      v16bf kb2 = load_b_frag(Kb, HEAD_DIM, 16, 0);
      v16bf kb3 = load_b_frag(Kb, HEAD_DIM, 16, 32);

#pragma unroll
      for (int t = 0; t < 2; ++t) {
        v8f s0, s1;
#pragma unroll
        for (int j = 0; j < 8; ++j) { s0[j] = 0.0f; s1[j] = 0.0f; }
        s0 = wmma_bf16(qa[t][0], kb0, s0);
        s0 = wmma_bf16(qa[t][1], kb1, s0);
        s1 = wmma_bf16(qa[t][0], kb2, s1);
        s1 = wmma_bf16(qa[t][1], kb3, s1);

        float blkmax[8];
#pragma unroll
        for (int j = 0; j < 8; ++j) {
          int qrow = q0 + 16 * t + j + 8 * hi;
          float v0 = s0[j] * scale; if (kv0 + n      > qrow) v0 = -INFINITY;
          float v1 = s1[j] * scale; if (kv0 + 16 + n > qrow) v1 = -INFINITY;
          s0[j] = v0; s1[j] = v1;
          blkmax[j] = fmaxf(v0, v1);
        }
#pragma unroll
        for (int j = 0; j < 8; ++j)
#pragma unroll
          for (int off = 8; off >= 1; off >>= 1)
            blkmax[j] = fmaxf(blkmax[j], __shfl_xor(blkmax[j], off, 16));

        float corr[8], psum[8];
#pragma unroll
        for (int j = 0; j < 8; ++j) {
          float mnew = fmaxf(rmax[t][j], blkmax[j]);
          corr[j] = __expf(rmax[t][j] - mnew);
          rmax[t][j] = mnew;
          float p0 = __expf(s0[j] - mnew);
          float p1 = __expf(s1[j] - mnew);
          s0[j] = p0; s1[j] = p1;
          psum[j] = p0 + p1;
        }
#pragma unroll
        for (int j = 0; j < 8; ++j)
#pragma unroll
          for (int off = 8; off >= 1; off >>= 1)
            psum[j] += __shfl_xor(psum[j], off, 16);
#pragma unroll
        for (int j = 0; j < 8; ++j) rsum[t][j] = rsum[t][j] * corr[j] + psum[j];
#pragma unroll
        for (int c = 0; c < 4; ++c)
#pragma unroll
          for (int j = 0; j < 8; ++j) acc[t][c][j] *= corr[j];

        // P re-layout via this wave's private LDS region (same-wave DS: in order)
        wave_fence();
#pragma unroll
        for (int j = 0; j < 8; ++j) {
          int row = j + 8 * hi;
          Pl[w][row * 32 + n]      = (__bf16)s0[j];
          Pl[w][row * 32 + 16 + n] = (__bf16)s1[j];
        }
        wave_fence();
        FragU pf;
        const __bf16* pp = &Pl[w][m * 32 + kb];
        pf.h[0] = *reinterpret_cast<const v8bf*>(pp);
        pf.h[1] = *reinterpret_cast<const v8bf*>(pp + 16);
        wave_fence();

#pragma unroll
        for (int c = 0; c < 4; ++c) {
          v16bf vb = load_b_frag(Vb, 32, c * 16, 0);
          acc[t][c] = wmma_bf16(pf.b, vb, acc[t][c]);
        }
      }
    }
    __syncthreads();                        // all reads done before re-staging buf
  }

  // ---- epilogue: normalize, write bf16 to attn [B, L, H*64]
  const int b  = bh >> 4;
  const int hh = bh & 15;
#pragma unroll
  for (int t = 0; t < 2; ++t) {
    float rinv[8];
#pragma unroll
    for (int j = 0; j < 8; ++j) rinv[j] = 1.0f / rsum[t][j];
#pragma unroll
    for (int c = 0; c < 4; ++c) {
#pragma unroll
      for (int j = 0; j < 8; ++j) {
        int q = q0 + 16 * t + j + 8 * hi;
        int d = c * 16 + n;
        attn[((size_t)b * SEQ + q) * N_EMBD + hh * HEAD_DIM + d] =
            (__bf16)(acc[t][c][j] * rinv[j]);
      }
    }
  }
}

// ---------------------------------------------------------------- output projection

__global__ __launch_bounds__(32)
void out_proj(const __bf16* __restrict__ attnb,
              const __bf16* __restrict__ Wot,
              const float* __restrict__ bo,
              float* __restrict__ out)
{
  const int row0 = blockIdx.y * 64;
  const int col0 = blockIdx.x * 64;

  v8f acc[4][4];
#pragma unroll
  for (int mi = 0; mi < 4; ++mi)
#pragma unroll
    for (int ni = 0; ni < 4; ++ni)
#pragma unroll
      for (int j = 0; j < 8; ++j) acc[mi][ni][j] = 0.0f;

  gemm_tile_64x64(attnb, N_EMBD, row0, Wot, N_EMBD, col0, N_EMBD, acc);

  const int l  = threadIdx.x & 31;
  const int n  = l & 15;
  const int hi = (l >= 16) ? 1 : 0;

  float biasv[4];
#pragma unroll
  for (int ni = 0; ni < 4; ++ni) biasv[ni] = bo[col0 + 16 * ni + n];

#pragma unroll
  for (int mi = 0; mi < 4; ++mi) {
#pragma unroll
    for (int ni = 0; ni < 4; ++ni) {
#pragma unroll
      for (int j = 0; j < 8; ++j) {
        int r = row0 + 16 * mi + j + 8 * hi;
        int c = col0 + 16 * ni + n;
        out[(size_t)r * N_EMBD + c] = acc[mi][ni][j] + biasv[ni];
      }
    }
  }
}

// ---------------------------------------------------------------- launch

extern "C" void kernel_launch(void* const* d_in, const int* in_sizes, int n_in,
                              void* d_out, int out_size, void* d_ws, size_t ws_size,
                              hipStream_t stream) {
  const float* x  = (const float*)d_in[0];
  const float* Wq = (const float*)d_in[1];
  const float* bq = (const float*)d_in[2];
  const float* Wk = (const float*)d_in[3];
  const float* bk = (const float*)d_in[4];
  const float* Wv = (const float*)d_in[5];
  const float* bv = (const float*)d_in[6];
  const float* Wo = (const float*)d_in[7];
  const float* bo = (const float*)d_in[8];
  float* out = (float*)d_out;

  char* ws = (char*)d_ws;
  __bf16* xb  = (__bf16*)(ws);                        // 16 MB
  __bf16* Wqt = (__bf16*)(ws + (size_t)16 * 1048576); // 2 MB each
  __bf16* Wkt = Wqt + (1u << 20);
  __bf16* Wvt = Wkt + (1u << 20);
  __bf16* Wot = Wvt + (1u << 20);
  __bf16* Qb  = (__bf16*)(ws + (size_t)24 * 1048576); // 16 MB
  __bf16* Kb  = Qb  + (size_t)ROWS * N_EMBD;          // 16 MB
  __bf16* Vtb = Kb  + (size_t)ROWS * N_EMBD;          // 16 MB
  __bf16* Ab  = Vtb + (size_t)ROWS * N_EMBD;          // 16 MB

  // 1) precision converts
  cvt_f32_to_bf16<<<(ROWS * N_EMBD) / (256 * 8), 256, 0, stream>>>(x, xb);
  {
    dim3 tgrid(N_EMBD / 32, N_EMBD / 32);
    transpose_cvt_w<<<tgrid, 256, 0, stream>>>(Wq, Wqt);
    transpose_cvt_w<<<tgrid, 256, 0, stream>>>(Wk, Wkt);
    transpose_cvt_w<<<tgrid, 256, 0, stream>>>(Wv, Wvt);
    transpose_cvt_w<<<tgrid, 256, 0, stream>>>(Wo, Wot);
  }

  // 2) fused QKV projection (WMMA, 64x64 tiles)
  {
    dim3 grid(N_EMBD / 64, ROWS / 64, 3);
    qkv_proj<<<grid, 32, 0, stream>>>(xb, Wqt, Wkt, Wvt, bq, bk, bv, Qb, Kb, Vtb);
  }

  // 3) flash attention (WMMA + async-LDS staged K/V, 4 waves/WG)
  {
    dim3 grid(SEQ / (32 * FA_WAVES), BATCH * N_HEAD);
    flash_attn<<<grid, 32 * FA_WAVES, 0, stream>>>(Qb, Kb, Vtb, Ab);
  }

  // 4) output projection (WMMA, f32 out + bias)
  {
    dim3 grid(N_EMBD / 64, ROWS / 64);
    out_proj<<<grid, 32, 0, stream>>>(Ab, Wot, bo, out);
  }
}
